// QueryCentricEncoder_74225624809624
// MI455X (gfx1250) — compile-verified, matrix operand order checked by reading, not run on hardware
//
#include <hip/hip_runtime.h>
#include <math.h>

// ---------------- dimensions ----------------
#define BN 16
#define AN 128
#define TN 21
#define MN 512
#define DD 256
#define HN 8
#define KNBR 32
#define FFD 1024
#define NLAYER 2
#define NEGC (-1000000000.0f)

// ---------------- vector types ----------------
typedef __bf16 v16bf __attribute__((ext_vector_type(16)));
typedef __bf16 v8bf  __attribute__((ext_vector_type(8)));
typedef float  v8f   __attribute__((ext_vector_type(8)));
typedef float  v4f   __attribute__((ext_vector_type(4)));

// ---------------- device helpers ----------------
__device__ __forceinline__ float wred_sum(float v) {
  #pragma unroll
  for (int m = 16; m >= 1; m >>= 1) v += __shfl_xor(v, m, 32);
  return v;
}
__device__ __forceinline__ float wred_max(float v) {
  #pragma unroll
  for (int m = 16; m >= 1; m >>= 1) v = fmaxf(v, __shfl_xor(v, m, 32));
  return v;
}

__device__ __forceinline__ v8f wmma_bf16(v16bf a, v16bf b, v8f c) {
  // v_wmma_f32_16x16x32_bf16: D = A(16x32) * B(32x16) + C(16x16 f32)
  return __builtin_amdgcn_wmma_f32_16x16x32_bf16(false, a, false, b, (short)0, c,
                                                 false, false);
}

__device__ __forceinline__ v16bf load_bf16_frag(const __bf16* row, int k0, int k1) {
  v8bf lo = *(const v8bf*)(row + k0);
  v8bf hi = *(const v8bf*)(row + k1);
  return __builtin_shufflevector(lo, hi, 0,1,2,3,4,5,6,7,8,9,10,11,12,13,14,15);
}

// ---------------- weight prep: f32 (K,N) -> bf16 transposed (N,K) ----------------
__global__ void transpose_to_bf16_kernel(const float* __restrict__ W, __bf16* __restrict__ Wt,
                                         int K, int N) {
  int i = blockIdx.x * 256 + threadIdx.x;
  if (i >= K * N) return;
  int n = i / K, k = i - n * K;
  Wt[(size_t)n * K + k] = (__bf16)W[(size_t)k * N + n];
}

// ---------------- generic bf16 WMMA GEMM ----------------
// result = X[rows,K](bf16) @ Wt^T + bias ; Wt stored [N][K] bf16.
// Block-cooperative: all 8 waves of a block share one 16-row A panel, staged
// into LDS via gfx1250 async global->LDS copies (ASYNCcnt). Each wave then
// computes a 16x32 output tile (A fragment from LDS reused for 2 WMMAs).
// flags: bit0 = relu, bit1 = Yf += result (residual, f32 path only).
// Exactly one of Yf / Ybf is non-null. Requires (N/32) % 8 == 0 or N/32 == 8.
__global__ void gemm_bf16_wmma_kernel(const __bf16* __restrict__ X, const __bf16* __restrict__ Wt,
                                      const float* __restrict__ bias,
                                      float* __restrict__ Yf, __bf16* __restrict__ Ybf,
                                      int rows, int N, int Kd, int flags) {
  extern __shared__ char smem[];
  __bf16* alds = (__bf16*)smem;          // 16 * Kd bf16
  int ntile = N >> 5;                    // 32-wide N tiles
  int nblk = ntile >> 3;                 // blocks per row-panel (1 for N=256, 4 for N=1024)
  int tm = blockIdx.x / nblk;
  int tnbase = (blockIdx.x - tm * nblk) * 8;
  if (tm >= (rows >> 4)) return;         // block-uniform
  int tid = threadIdx.x;

  // ---- async stage: X[tm*16 .. tm*16+15][0..Kd) -> LDS (16B chunks/lane) ----
  {
    unsigned long long gbase = (unsigned long long)(const void*)(X + (size_t)tm * 16 * Kd);
    unsigned ldsbase = (unsigned)(size_t)(void*)alds;  // low 32 bits = LDS byte address
    int nchunk = (16 * Kd) / 8;          // #16-byte chunks (Kd*2 bytes per row)
    for (int c = tid; c < nchunk; c += 256) {
      unsigned voff = (unsigned)c * 16u;
      unsigned laddr = ldsbase + voff;
      asm volatile("global_load_async_to_lds_b128 %0, %1, %2"
                   :
                   : "v"(laddr), "v"(voff), "s"(gbase)
                   : "memory");
    }
    asm volatile("s_wait_asynccnt 0x0" ::: "memory");
    __syncthreads();
  }

  int w = tid >> 5, lane = tid & 31, half = lane >> 4, r = lane & 15;
  int tn = tnbase + w;
  const __bf16* ar  = alds + (size_t)r * Kd;
  const __bf16* wr0 = Wt + (size_t)(tn * 32 + r) * Kd;
  const __bf16* wr1 = wr0 + (size_t)16 * Kd;
  v8f acc0 = {0.f, 0.f, 0.f, 0.f, 0.f, 0.f, 0.f, 0.f};
  v8f acc1 = {0.f, 0.f, 0.f, 0.f, 0.f, 0.f, 0.f, 0.f};
  for (int kb = 0; kb < Kd; kb += 32) {
    int k0 = kb + half * 8, k1 = kb + 16 + half * 8;
    __builtin_prefetch(wr0 + kb + 64, 0, 0);  // global_prefetch_b8
    v16bf av  = load_bf16_frag(ar,  k0, k1);  // ds_load_b128 x2
    v16bf bv0 = load_bf16_frag(wr0, k0, k1);
    v16bf bv1 = load_bf16_frag(wr1, k0, k1);
    acc0 = wmma_bf16(av, bv0, acc0);
    acc1 = wmma_bf16(av, bv1, acc1);
  }
  int col0 = tn * 32 + r, col1 = col0 + 16;
  float bs0 = bias ? bias[col0] : 0.f;
  float bs1 = bias ? bias[col1] : 0.f;
  #pragma unroll
  for (int v = 0; v < 8; ++v) {
    int m = tm * 16 + half * 8 + v;
    float v0 = acc0[v] + bs0;
    float v1 = acc1[v] + bs1;
    if (flags & 1) { v0 = fmaxf(v0, 0.f); v1 = fmaxf(v1, 0.f); }
    if (Yf) {
      float* yp0 = Yf + (size_t)m * N + col0;
      float* yp1 = Yf + (size_t)m * N + col1;
      if (flags & 2) { v0 += *yp0; v1 += *yp1; }
      *yp0 = v0; *yp1 = v1;
    } else {
      Ybf[(size_t)m * N + col0] = (__bf16)v0;
      Ybf[(size_t)m * N + col1] = (__bf16)v1;
    }
  }
}

// ---------------- layernorm: one wave per row of 256, bf16 output ----------------
__global__ void ln_kernel(const float* __restrict__ x, const float* __restrict__ g,
                          const float* __restrict__ bt, __bf16* __restrict__ y, int rows) {
  int w = threadIdx.x >> 5, lane = threadIdx.x & 31;
  int row = blockIdx.x * 8 + w;
  if (row >= rows) return;
  const float* xr = x + (size_t)row * DD;
  float v[8], s = 0.f;
  #pragma unroll
  for (int i = 0; i < 8; ++i) { v[i] = xr[i * 32 + lane]; s += v[i]; }
  s = wred_sum(s);
  float mean = s * (1.f / DD);
  float q = 0.f;
  #pragma unroll
  for (int i = 0; i < 8; ++i) { v[i] -= mean; q += v[i] * v[i]; }
  q = wred_sum(q);
  float rs = rsqrtf(q * (1.f / DD) + 1e-5f);
  #pragma unroll
  for (int i = 0; i < 8; ++i) {
    int c = i * 32 + lane;
    y[(size_t)row * DD + c] = (__bf16)(v[i] * rs * g[c] + bt[c]);
  }
}

__global__ void final_ln_mask_kernel(const float* __restrict__ x, const float* __restrict__ g,
                                     const float* __restrict__ bt,
                                     const unsigned char* __restrict__ ctrl,
                                     const unsigned char* __restrict__ am,
                                     float* __restrict__ out, int rows) {
  int w = threadIdx.x >> 5, lane = threadIdx.x & 31;
  int row = blockIdx.x * 8 + w;
  if (row >= rows) return;
  const float* xr = x + (size_t)row * DD;
  float v[8], s = 0.f;
  #pragma unroll
  for (int i = 0; i < 8; ++i) { v[i] = xr[i * 32 + lane]; s += v[i]; }
  s = wred_sum(s);
  float mean = s * (1.f / DD);
  float q = 0.f;
  #pragma unroll
  for (int i = 0; i < 8; ++i) { v[i] -= mean; q += v[i] * v[i]; }
  q = wred_sum(q);
  float rs = rsqrtf(q * (1.f / DD) + 1e-5f);
  float msk = (ctrl[row] && am[row]) ? 1.f : 0.f;
  #pragma unroll
  for (int i = 0; i < 8; ++i) {
    int c = i * 32 + lane;
    out[(size_t)row * DD + c] = (v[i] * rs * g[c] + bt[c]) * msk;
  }
}

__global__ void mask_rows_kernel(float* __restrict__ x, const unsigned char* __restrict__ m,
                                 int rows) {
  int i = blockIdx.x * 256 + threadIdx.x;
  if (i >= rows * DD) return;
  if (!m[i >> 8]) x[i] = 0.f;
}

// ---------------- top-k neighbor selection ----------------
__global__ void topk_kernel(const float* __restrict__ qpos, const float* __restrict__ kpos,
                            const unsigned char* __restrict__ kmask, int* __restrict__ out,
                            int Nq, int Nkv, int total) {
  int g = blockIdx.x * 256 + threadIdx.x;
  if (g >= total) return;
  int b = g / Nq, n = g - b * Nq;
  float qx = qpos[(size_t)(b * Nq + n) * 2];
  float qy = qpos[(size_t)(b * Nq + n) * 2 + 1];
  float best[KNBR];
  int   bi[KNBR];
  #pragma unroll
  for (int i = 0; i < KNBR; ++i) { best[i] = INFINITY; bi[i] = i; }
  for (int c = 0; c < Nkv; ++c) {
    float dx = kpos[(size_t)(b * Nkv + c) * 2] - qx;
    float dy = kpos[(size_t)(b * Nkv + c) * 2 + 1] - qy;
    float d2 = kmask[b * Nkv + c] ? (dx * dx + dy * dy) : INFINITY;
    if (d2 < best[KNBR - 1]) {
      best[KNBR - 1] = d2; bi[KNBR - 1] = c;
      #pragma unroll
      for (int s = KNBR - 1; s > 0; --s) {
        if (best[s] < best[s - 1]) {
          float tf = best[s]; best[s] = best[s - 1]; best[s - 1] = tf;
          int ti = bi[s]; bi[s] = bi[s - 1]; bi[s - 1] = ti;
        }
      }
    }
  }
  #pragma unroll
  for (int i = 0; i < KNBR; ++i) out[(size_t)(b * Nq + n) * KNBR + i] = bi[i];
}

// ---------------- sparse attention core ----------------
__global__ void attn_core_kernel(const float* __restrict__ q, const float* __restrict__ kbase,
                                 const float* __restrict__ vbase, const int* __restrict__ idx,
                                 const float* __restrict__ qpos, const float* __restrict__ kvpos,
                                 const float* __restrict__ qhd, const float* __restrict__ kvhd,
                                 const unsigned char* __restrict__ kvmask,
                                 const float* __restrict__ Wrk, const float* __restrict__ Wrv,
                                 __bf16* __restrict__ out, int Nq, int Nkv) {
  __shared__ float  qs[DD];
  __shared__ __bf16 ks[KNBR][DD];
  __shared__ __bf16 vs[KNBR][DD];
  __shared__ float  rpe[KNBR][5];
  __shared__ int    nbr[KNBR];
  __shared__ float  mgs[KNBR];
  int bn = blockIdx.x;
  int b = bn / Nq;
  int tid = threadIdx.x;
  if (tid < KNBR) {
    int j = tid;
    int ij = idx[(size_t)bn * KNBR + j];
    nbr[j] = ij;
    mgs[j] = kvmask[b * Nkv + ij] ? 1.f : 0.f;
    float qx = qpos[(size_t)bn * 2], qy = qpos[(size_t)bn * 2 + 1];
    float kx = kvpos[(size_t)(b * Nkv + ij) * 2], ky = kvpos[(size_t)(b * Nkv + ij) * 2 + 1];
    float h = qhd[bn];
    float c = cosf(h), s = sinf(h);
    float dx = kx - qx, dy = ky - qy;
    float rx = c * dx + s * dy, ry = -s * dx + c * dy;
    float rh = kvhd[b * Nkv + ij] - h;
    rpe[j][0] = rx; rpe[j][1] = ry; rpe[j][2] = sqrtf(rx * rx + ry * ry);
    rpe[j][3] = sinf(rh); rpe[j][4] = cosf(rh);
  }
  qs[tid] = q[(size_t)bn * DD + tid];
  __syncthreads();
  {
    int d = tid;
    for (int j = 0; j < KNBR; ++j) {
      int ij = nbr[j];
      float rk = 0.f, rv = 0.f;
      #pragma unroll
      for (int c = 0; c < 5; ++c) {
        float rc = rpe[j][c];
        rk += rc * Wrk[c * DD + d];
        rv += rc * Wrv[c * DD + d];
      }
      ks[j][d] = (__bf16)(kbase[((size_t)b * Nkv + ij) * DD + d] + rk);
      vs[j][d] = (__bf16)(vbase[((size_t)b * Nkv + ij) * DD + d] + rv);
    }
  }
  __syncthreads();
  int h = tid >> 5, lane = tid & 31;
  float sc = 0.f;
  #pragma unroll
  for (int dd = 0; dd < 32; ++dd) sc += qs[h * 32 + dd] * (float)ks[lane][h * 32 + dd];
  sc *= 0.17677669529663687f;  // 1/sqrt(32)
  if (mgs[lane] == 0.f) sc = NEGC;
  float mx = wred_max(sc);
  float e = __expf(sc - mx);
  float ssum = wred_sum(e);
  float a = e / ssum;
  float acc = 0.f;
  #pragma unroll
  for (int j = 0; j < KNBR; ++j) {
    float aj = __shfl(a, j, 32);
    acc += aj * (float)vs[j][h * 32 + lane];
  }
  out[(size_t)bn * DD + h * 32 + lane] = (__bf16)acc;
}

// ---------------- fused agent encoder ----------------
__global__ void agent_encoder_kernel(
    const float* __restrict__ trajs, const float* __restrict__ pos, const float* __restrict__ head,
    const unsigned char* __restrict__ tmask, const unsigned char* __restrict__ amask,
    const int* __restrict__ types, const unsigned char* __restrict__ ctrl,
    const int* __restrict__ sdc_idx,
    const float* __restrict__ W1, const float* __restrict__ b1,
    const float* __restrict__ g1, const float* __restrict__ n1,
    const __bf16* __restrict__ W2t, const float* __restrict__ b2,
    const float* __restrict__ go, const float* __restrict__ no,
    float* __restrict__ agent_feat) {
  __shared__ float  fs[TN][20];
  __shared__ float  hbuf[32][DD];
  __shared__ __bf16 abuf[32][DD];
  __shared__ float  red[DD];
  int ba = blockIdx.x;
  int b = ba / AN, a = ba - b * AN;
  int tid = threadIdx.x;
  float px = pos[(size_t)ba * 2], py = pos[(size_t)ba * 2 + 1];
  float hh = head[ba];
  float ch = cosf(hh), sh = sinf(hh);
  if (tid < TN) {
    int t = tid;
    const float* tr  = trajs + ((size_t)ba * TN + t) * 10;
    const float* trp = trajs + ((size_t)ba * TN + (t > 0 ? t - 1 : 0)) * 10;
    float x = tr[0] - px, y = tr[1] - py;
    float lx = ch * x + sh * y, ly = -sh * x + ch * y;
    float lvx = ch * tr[8] + sh * tr[9], lvy = -sh * tr[8] + ch * tr[9];
    float plvx = ch * trp[8] + sh * trp[9], plvy = -sh * trp[8] + ch * trp[9];
    float ax = (lvx - plvx) * 10.f, ay = (lvy - plvy) * 10.f;
    float rh = atan2f(tr[6], tr[7]) - hh;
    int ty = types[ba];
    int sdc = sdc_idx[b];
    float msk = tmask[(size_t)ba * TN + t] ? 1.f : 0.f;
    float f[20];
    f[0] = lx; f[1] = ly; f[2] = tr[2]; f[3] = tr[3]; f[4] = tr[4]; f[5] = tr[5];
    f[6] = sinf(rh); f[7] = cosf(rh); f[8] = lvx; f[9] = lvy; f[10] = ax; f[11] = ay;
    f[12] = (ty == 1) ? 1.f : 0.f;
    f[13] = (ty == 2) ? 1.f : 0.f;
    f[14] = (ty == 3) ? 1.f : 0.f;
    f[15] = ((ty <= 0) || (ty > 3)) ? 1.f : 0.f;
    f[16] = ctrl[ba] ? 1.f : 0.f;
    f[17] = (sdc >= 0 && sdc < AN && sdc == a) ? 1.f : 0.f;
    f[18] = -1.f + (float)t * (1.f / 20.f);
    #pragma unroll
    for (int c = 0; c < 19; ++c) fs[t][c] = msk ? f[c] : 0.f;
    fs[t][19] = msk;
  }
  __syncthreads();
  { // W1 (K=20, VALU)
    int col = tid;
    for (int t = 0; t < TN; ++t) {
      float s = b1[col];
      #pragma unroll
      for (int c = 0; c < 20; ++c) s += fs[t][c] * W1[c * DD + col];
      hbuf[t][col] = s;
    }
  }
  __syncthreads();
  { // LN(g1,n1) + relu -> bf16 staging (rows 21..31 zero)
    int w = tid >> 5, lane = tid & 31;
    for (int rr = w; rr < 32; rr += 8) {
      if (rr < TN) {
        float v[8], s = 0.f;
        #pragma unroll
        for (int i = 0; i < 8; ++i) { v[i] = hbuf[rr][i * 32 + lane]; s += v[i]; }
        s = wred_sum(s);
        float mean = s * (1.f / DD), qv = 0.f;
        #pragma unroll
        for (int i = 0; i < 8; ++i) { v[i] -= mean; qv += v[i] * v[i]; }
        qv = wred_sum(qv);
        float rs = rsqrtf(qv * (1.f / DD) + 1e-5f);
        #pragma unroll
        for (int i = 0; i < 8; ++i) {
          int c = i * 32 + lane;
          abuf[rr][c] = (__bf16)fmaxf(v[i] * rs * g1[c] + n1[c], 0.f);
        }
      } else {
        #pragma unroll
        for (int i = 0; i < 8; ++i) abuf[rr][i * 32 + lane] = (__bf16)0.f;
      }
    }
  }
  __syncthreads();
  { // W2 via WMMA: 32x256 @ 256x256, 32 tiles over 8 waves
    int w = tid >> 5, lane = tid & 31, half = lane >> 4, r = lane & 15;
    for (int tile = w; tile < 32; tile += 8) {
      int tm = tile >> 4, tn = tile & 15;
      v8f acc = {0.f, 0.f, 0.f, 0.f, 0.f, 0.f, 0.f, 0.f};
      const __bf16* arow = &abuf[tm * 16 + r][0];
      const __bf16* wr = W2t + (size_t)(tn * 16 + r) * DD;
      for (int kb = 0; kb < DD; kb += 32) {
        int k0 = kb + half * 8, k1 = kb + 16 + half * 8;
        acc = wmma_bf16(load_bf16_frag(arow, k0, k1), load_bf16_frag(wr, k0, k1), acc);
      }
      int col = tn * 16 + r;
      #pragma unroll
      for (int v = 0; v < 8; ++v) hbuf[tm * 16 + half * 8 + v][col] = acc[v] + b2[col];
    }
  }
  __syncthreads();
  { // masked max over T, then block LN(go,no), mask, write
    int col = tid;
    float m = NEGC;
    for (int t = 0; t < TN; ++t) {
      float val = tmask[(size_t)ba * TN + t] ? hbuf[t][col] : NEGC;
      m = fmaxf(m, val);
    }
    red[col] = m; __syncthreads();
    for (int s2 = 128; s2 > 0; s2 >>= 1) { if (col < s2) red[col] += red[col + s2]; __syncthreads(); }
    float mean = red[0] * (1.f / DD);
    __syncthreads();
    float dcen = m - mean;
    red[col] = dcen * dcen; __syncthreads();
    for (int s2 = 128; s2 > 0; s2 >>= 1) { if (col < s2) red[col] += red[col + s2]; __syncthreads(); }
    float rs = rsqrtf(red[0] * (1.f / DD) + 1e-5f);
    float af = dcen * rs * go[col] + no[col];
    af = amask[ba] ? af : 0.f;
    agent_feat[(size_t)ba * DD + col] = af;
  }
}

// ---------------- fused map encoder (16 tokens per block) ----------------
__global__ void map_encoder_kernel(const float* __restrict__ tok,
                                   const unsigned char* __restrict__ mmask,
                                   const float* __restrict__ W1, const float* __restrict__ b1,
                                   const float* __restrict__ g1, const float* __restrict__ n1,
                                   const __bf16* __restrict__ W2t, const float* __restrict__ b2,
                                   const float* __restrict__ g2, const float* __restrict__ n2,
                                   float* __restrict__ map_feat) {
  __shared__ float  tf[16][12];
  __shared__ float  hbuf[16][DD];
  __shared__ __bf16 abuf[16][DD];
  int blk = blockIdx.x;
  int b = blk >> 5, m0 = (blk & 31) * 16;
  int tid = threadIdx.x;
  if (tid < 16) {
    const float* t = tok + ((size_t)b * MN + m0 + tid) * 11;
    float c0 = t[0], c1 = t[1];
    tf[tid][0] = 0.f; tf[tid][1] = 0.f;
    tf[tid][2] = t[2] - c0; tf[tid][3] = t[3] - c1;
    tf[tid][4] = t[4] - c0; tf[tid][5] = t[5] - c1;
    tf[tid][6] = t[6]; tf[tid][7] = t[7];
    tf[tid][8] = t[8]; tf[tid][9] = t[9]; tf[tid][10] = t[10];
  }
  __syncthreads();
  {
    int col = tid;
    for (int rr = 0; rr < 16; ++rr) {
      float s = b1[col];
      #pragma unroll
      for (int c = 0; c < 11; ++c) s += tf[rr][c] * W1[c * DD + col];
      hbuf[rr][col] = s;
    }
  }
  __syncthreads();
  { // LN(g1,n1)+relu -> bf16
    int w = tid >> 5, lane = tid & 31;
    for (int rr = w; rr < 16; rr += 8) {
      float v[8], s = 0.f;
      #pragma unroll
      for (int i = 0; i < 8; ++i) { v[i] = hbuf[rr][i * 32 + lane]; s += v[i]; }
      s = wred_sum(s);
      float mean = s * (1.f / DD), qv = 0.f;
      #pragma unroll
      for (int i = 0; i < 8; ++i) { v[i] -= mean; qv += v[i] * v[i]; }
      qv = wred_sum(qv);
      float rs = rsqrtf(qv * (1.f / DD) + 1e-5f);
      #pragma unroll
      for (int i = 0; i < 8; ++i) {
        int c = i * 32 + lane;
        abuf[rr][c] = (__bf16)fmaxf(v[i] * rs * g1[c] + n1[c], 0.f);
      }
    }
  }
  __syncthreads();
  { // WMMA 16x256 @ 256x256 (tm=0), 16 col tiles over 8 waves
    int w = tid >> 5, lane = tid & 31, half = lane >> 4, r = lane & 15;
    for (int tn = w; tn < 16; tn += 8) {
      v8f acc = {0.f, 0.f, 0.f, 0.f, 0.f, 0.f, 0.f, 0.f};
      const __bf16* arow = &abuf[r][0];
      const __bf16* wr = W2t + (size_t)(tn * 16 + r) * DD;
      for (int kb = 0; kb < DD; kb += 32) {
        int k0 = kb + half * 8, k1 = kb + 16 + half * 8;
        acc = wmma_bf16(load_bf16_frag(arow, k0, k1), load_bf16_frag(wr, k0, k1), acc);
      }
      int col = tn * 16 + r;
      #pragma unroll
      for (int v = 0; v < 8; ++v) hbuf[half * 8 + v][col] = acc[v] + b2[col];
    }
  }
  __syncthreads();
  { // LN(g2,n2), mask, write
    int w = tid >> 5, lane = tid & 31;
    for (int rr = w; rr < 16; rr += 8) {
      float v[8], s = 0.f;
      #pragma unroll
      for (int i = 0; i < 8; ++i) { v[i] = hbuf[rr][i * 32 + lane]; s += v[i]; }
      s = wred_sum(s);
      float mean = s * (1.f / DD), qv = 0.f;
      #pragma unroll
      for (int i = 0; i < 8; ++i) { v[i] -= mean; qv += v[i] * v[i]; }
      qv = wred_sum(qv);
      float rs = rsqrtf(qv * (1.f / DD) + 1e-5f);
      float msk = mmask[b * MN + m0 + rr] ? 1.f : 0.f;
      #pragma unroll
      for (int i = 0; i < 8; ++i) {
        int c = i * 32 + lane;
        map_feat[((size_t)b * MN + m0 + rr) * DD + c] = (v[i] * rs * g2[c] + n2[c]) * msk;
      }
    }
  }
}

// ---------------- goal MLP hidden (K=5, VALU), bf16 output ----------------
__global__ void goal_hidden_kernel(const float* __restrict__ goalp, const float* __restrict__ pos,
                                   const float* __restrict__ head, const float* __restrict__ W1,
                                   const float* __restrict__ b1, __bf16* __restrict__ out) {
  int ba = blockIdx.x, col = threadIdx.x;
  float gx = goalp[(size_t)ba * 2] - pos[(size_t)ba * 2];
  float gy = goalp[(size_t)ba * 2 + 1] - pos[(size_t)ba * 2 + 1];
  float h = head[ba], c = cosf(h), s = sinf(h);
  float lx = c * gx + s * gy, ly = -s * gx + c * gy;
  float d = sqrtf(lx * lx + ly * ly);
  float dm = fmaxf(d, 0.001f);
  float f0 = lx, f1 = ly, f2 = d, f3 = lx / dm, f4 = ly / dm;
  float acc = b1[col] + f0 * W1[col] + f1 * W1[DD + col] + f2 * W1[2 * DD + col] +
              f3 * W1[3 * DD + col] + f4 * W1[4 * DD + col];
  out[(size_t)ba * DD + col] = (__bf16)fmaxf(acc, 0.f);
}

// ---------------- host orchestration ----------------
namespace {
struct AttnP { const float *Wk, *Wo, *Wq, *Wrk, *Wrv, *Wv, *bk, *bo, *bq, *bv; };
struct FfnP  { const float *W1, *W2, *b1, *b2, *g, *n; };
struct LnP   { const float *b, *g; };
struct LayerP {
  AttnP aa, am, mm;
  FfnP fa, fm;
  LnP ln_aa, ln_am, ln_am_kv, ln_mm;
};
struct AttnT { __bf16 *q, *k, *v, *o; };
struct FfnT  { __bf16 *w1, *w2; };
}  // namespace

extern "C" void kernel_launch(void* const* d_in, const int* in_sizes, int n_in,
                              void* d_out, int out_size, void* d_ws, size_t ws_size,
                              hipStream_t stream) {
  (void)in_sizes; (void)n_in; (void)out_size; (void)ws_size;
  const float* obj_trajs      = (const float*)d_in[0];
  const float* obj_positions  = (const float*)d_in[1];
  const float* obj_headings   = (const float*)d_in[2];
  const float* map_tok        = (const float*)d_in[3];
  const float* map_centers    = (const float*)d_in[4];
  const float* map_headings   = (const float*)d_in[5];
  const float* goal_positions = (const float*)d_in[6];
  const float* P              = (const float*)d_in[7];
  const unsigned char* tmask      = (const unsigned char*)d_in[8];
  const unsigned char* agent_mask = (const unsigned char*)d_in[9];
  const unsigned char* map_mask   = (const unsigned char*)d_in[10];
  const int* obj_types            = (const int*)d_in[11];
  const unsigned char* ctrl_mask  = (const unsigned char*)d_in[12];
  const int* sdc_idx              = (const int*)d_in[13];

  // ---- params: jax pytree (dicts sorted by key) ----
  size_t off = 0;
  auto F = [&](size_t n) { const float* q = P + off; off += n; return q; };
  // agent_enc: W1 W2 b1 b2 g1 go n1 no
  const float* ae_W1 = F(20 * DD); const float* ae_W2 = F(DD * DD);
  const float* ae_b1 = F(DD); const float* ae_b2 = F(DD);
  const float* ae_g1 = F(DD); const float* ae_go = F(DD);
  const float* ae_n1 = F(DD); const float* ae_no = F(DD);
  // goal: W1 W2 b1 b2
  const float* gl_W1 = F(5 * DD); const float* gl_W2 = F(DD * DD);
  const float* gl_b1 = F(DD); const float* gl_b2 = F(DD);
  // layers: aa am ffn_a ffn_m ln_aa ln_am ln_am_kv ln_mm mm
  auto getAttn = [&]() {
    AttnP t;
    t.Wk = F(DD * DD); t.Wo = F(DD * DD); t.Wq = F(DD * DD);
    t.Wrk = F(5 * DD); t.Wrv = F(5 * DD); t.Wv = F(DD * DD);
    t.bk = F(DD); t.bo = F(DD); t.bq = F(DD); t.bv = F(DD);
    return t;
  };
  auto getFfn = [&]() {
    FfnP t;
    t.W1 = F(DD * FFD); t.W2 = F(FFD * DD);
    t.b1 = F(FFD); t.b2 = F(DD); t.g = F(DD); t.n = F(DD);
    return t;
  };
  auto getLn = [&]() { LnP t; t.b = F(DD); t.g = F(DD); return t; };
  LayerP LY[NLAYER];
  for (int l = 0; l < NLAYER; ++l) {
    LY[l].aa = getAttn();
    LY[l].am = getAttn();
    LY[l].fa = getFfn();
    LY[l].fm = getFfn();
    LY[l].ln_aa = getLn();
    LY[l].ln_am = getLn();
    LY[l].ln_am_kv = getLn();
    LY[l].ln_mm = getLn();
    LY[l].mm = getAttn();
  }
  // map_enc: W1 W2 b1 b2 g1 g2 n1 n2
  const float* me_W1 = F(11 * DD); const float* me_W2 = F(DD * DD);
  const float* me_b1 = F(DD); const float* me_b2 = F(DD);
  const float* me_g1 = F(DD); const float* me_g2 = F(DD);
  const float* me_n1 = F(DD); const float* me_n2 = F(DD);
  // out_norm: b g
  const float* on_b = F(DD); const float* on_g = F(DD);

  // ---- workspace bump allocator ----
  char* base = (char*)d_ws;
  size_t wo = 0;
  auto alloc = [&](size_t bytes) {
    void* p = base + wo;
    wo = (wo + bytes + 255) & ~(size_t)255;
    return p;
  };
  auto makeT = [&](const float* W, int K, int N) {
    __bf16* t = (__bf16*)alloc((size_t)K * N * sizeof(__bf16));
    int total = K * N;
    transpose_to_bf16_kernel<<<(total + 255) / 256, 256, 0, stream>>>(W, t, K, N);
    return t;
  };

  // bf16 transposed weights
  __bf16* ae_W2t = makeT(ae_W2, DD, DD);
  __bf16* me_W2t = makeT(me_W2, DD, DD);
  __bf16* gl_W2t = makeT(gl_W2, DD, DD);
  AttnT Tmm[NLAYER], Taa[NLAYER], Tam[NLAYER];
  FfnT Tfa[NLAYER], Tfm[NLAYER];
  for (int l = 0; l < NLAYER; ++l) {
    Tmm[l] = {makeT(LY[l].mm.Wq, DD, DD), makeT(LY[l].mm.Wk, DD, DD),
              makeT(LY[l].mm.Wv, DD, DD), makeT(LY[l].mm.Wo, DD, DD)};
    Taa[l] = {makeT(LY[l].aa.Wq, DD, DD), makeT(LY[l].aa.Wk, DD, DD),
              makeT(LY[l].aa.Wv, DD, DD), makeT(LY[l].aa.Wo, DD, DD)};
    Tam[l] = {makeT(LY[l].am.Wq, DD, DD), makeT(LY[l].am.Wk, DD, DD),
              makeT(LY[l].am.Wv, DD, DD), makeT(LY[l].am.Wo, DD, DD)};
    Tfa[l] = {makeT(LY[l].fa.W1, DD, FFD), makeT(LY[l].fa.W2, FFD, DD)};
    Tfm[l] = {makeT(LY[l].fm.W1, DD, FFD), makeT(LY[l].fm.W2, FFD, DD)};
  }

  // activation buffers
  const int AR = BN * AN;   // 2048 agent rows
  const int MR = BN * MN;   // 8192 map rows
  float*  agent_feat = (float*)alloc((size_t)AR * DD * 4);
  float*  map_feat   = (float*)alloc((size_t)MR * DD * 4);
  __bf16* xa         = (__bf16*)alloc((size_t)AR * DD * 2);
  __bf16* xm         = (__bf16*)alloc((size_t)MR * DD * 2);
  // big region: during attention -> q/k/v (f32) + ao (bf16); during FFN -> hidden (bf16)
  char* big = (char*)alloc((size_t)MR * DD * 4 * 3 + (size_t)MR * DD * 2);
  float*  qb  = (float*)big;
  float*  kb  = (float*)(big + (size_t)MR * DD * 4);
  float*  vb  = (float*)(big + (size_t)MR * DD * 8);
  __bf16* ao  = (__bf16*)(big + (size_t)MR * DD * 12);
  __bf16* hid = (__bf16*)big;  // MR*FFD*2 bytes <= region size
  int* idx_mm = (int*)alloc((size_t)BN * MN * KNBR * 4);
  int* idx_aa = (int*)alloc((size_t)BN * AN * KNBR * 4);
  int* idx_am = (int*)alloc((size_t)BN * AN * KNBR * 4);

  auto gemm_f = [&](const __bf16* X, const __bf16* Wt, const float* bias, float* Y,
                    int rows, int N, int K, int flags) {
    int blocks = (rows / 16) * (N / 32 / 8);
    size_t sh = (size_t)16 * K * sizeof(__bf16);
    gemm_bf16_wmma_kernel<<<blocks, 256, sh, stream>>>(X, Wt, bias, Y, nullptr,
                                                       rows, N, K, flags);
  };
  auto gemm_b = [&](const __bf16* X, const __bf16* Wt, const float* bias, __bf16* Y,
                    int rows, int N, int K, int flags) {
    int blocks = (rows / 16) * (N / 32 / 8);
    size_t sh = (size_t)16 * K * sizeof(__bf16);
    gemm_bf16_wmma_kernel<<<blocks, 256, sh, stream>>>(X, Wt, bias, nullptr, Y,
                                                       rows, N, K, flags);
  };
  auto ln = [&](const float* X, const float* g, const float* bt, __bf16* Y, int rows) {
    ln_kernel<<<(rows + 7) / 8, 256, 0, stream>>>(X, g, bt, Y, rows);
  };
  auto attn = [&](const AttnP& ap, const AttnT& at, const __bf16* xq, const __bf16* xkv,
                  int Nq, int Nkv, const float* qp, const float* kp,
                  const float* qh, const float* kh, const unsigned char* km,
                  const int* idxp, float* Yres) {
    gemm_f(xq,  at.q, ap.bq, qb, BN * Nq, DD, DD, 0);
    gemm_f(xkv, at.k, ap.bk, kb, BN * Nkv, DD, DD, 0);
    gemm_f(xkv, at.v, ap.bv, vb, BN * Nkv, DD, DD, 0);
    attn_core_kernel<<<BN * Nq, 256, 0, stream>>>(qb, kb, vb, idxp, qp, kp, qh, kh, km,
                                                  ap.Wrk, ap.Wrv, ao, Nq, Nkv);
    gemm_f(ao, at.o, ap.bo, Yres, BN * Nq, DD, DD, 2);
  };
  auto ffn = [&](const FfnP& fp, const FfnT& ft, float* x, __bf16* xn, int rows) {
    ln(x, fp.g, fp.n, xn, rows);
    gemm_b(xn, ft.w1, fp.b1, hid, rows, FFD, DD, 1);
    gemm_f(hid, ft.w2, fp.b2, x, rows, DD, FFD, 2);
  };

  // ---- encoders ----
  agent_encoder_kernel<<<AR, 256, 0, stream>>>(
      obj_trajs, obj_positions, obj_headings, tmask, agent_mask, obj_types, ctrl_mask, sdc_idx,
      ae_W1, ae_b1, ae_g1, ae_n1, ae_W2t, ae_b2, ae_go, ae_no, agent_feat);
  map_encoder_kernel<<<BN * (MN / 16), 256, 0, stream>>>(
      map_tok, map_mask, me_W1, me_b1, me_g1, me_n1, me_W2t, me_b2, me_g2, me_n2, map_feat);

  // ---- neighbor selection ----
  topk_kernel<<<(BN * MN + 255) / 256, 256, 0, stream>>>(map_centers, map_centers, map_mask,
                                                         idx_mm, MN, MN, BN * MN);
  topk_kernel<<<(BN * AN + 255) / 256, 256, 0, stream>>>(obj_positions, obj_positions, agent_mask,
                                                         idx_aa, AN, AN, BN * AN);
  topk_kernel<<<(BN * AN + 255) / 256, 256, 0, stream>>>(obj_positions, map_centers, map_mask,
                                                         idx_am, AN, MN, BN * AN);

  // ---- transformer layers ----
  for (int l = 0; l < NLAYER; ++l) {
    const LayerP& L = LY[l];
    // map-map
    ln(map_feat, L.ln_mm.g, L.ln_mm.b, xm, MR);
    attn(L.mm, Tmm[l], xm, xm, MN, MN, map_centers, map_centers, map_headings, map_headings,
         map_mask, idx_mm, map_feat);
    // agent-agent
    ln(agent_feat, L.ln_aa.g, L.ln_aa.b, xa, AR);
    attn(L.aa, Taa[l], xa, xa, AN, AN, obj_positions, obj_positions, obj_headings, obj_headings,
         agent_mask, idx_aa, agent_feat);
    // agent-map (cross)
    ln(agent_feat, L.ln_am.g, L.ln_am.b, xa, AR);
    ln(map_feat, L.ln_am_kv.g, L.ln_am_kv.b, xm, MR);
    attn(L.am, Tam[l], xa, xm, AN, MN, obj_positions, map_centers, obj_headings, map_headings,
         map_mask, idx_am, agent_feat);
    // FFNs
    ffn(L.fa, Tfa[l], agent_feat, xa, AR);
    ffn(L.fm, Tfm[l], map_feat, xm, MR);
  }

  // ---- tail: mask, goal MLP, residual, output LN ----
  mask_rows_kernel<<<(AR * DD + 255) / 256, 256, 0, stream>>>(agent_feat, agent_mask, AR);
  goal_hidden_kernel<<<AR, 256, 0, stream>>>(goal_positions, obj_positions, obj_headings,
                                             gl_W1, gl_b1, xa);
  gemm_f(xa, gl_W2t, gl_b2, agent_feat, AR, DD, DD, 2);
  final_ln_mask_kernel<<<(AR + 7) / 8, 256, 0, stream>>>(agent_feat, on_g, on_b, ctrl_mask,
                                                         agent_mask, (float*)d_out, AR);
}
